// Decoder_78099685310770
// MI455X (gfx1250) — matrix operand order repaired and checked by reading, not compiled, not run on hardware
//
#include <hip/hip_runtime.h>
#include <hip/hip_bf16.h>

// ---------------------------------------------------------------------------
// MI455X (gfx1250) decoder pipeline.
// Dominant cost: Y = U2[16384,4096] @ X[4096,256]  (34.4 GFLOP, 256 MB fp32).
// Strategy: bf16x3 compensated WMMA GEMM (Ah*Bh + Ah*Bl + Al*Bh) using
// v_wmma_f32_16x16x32_bf16 -> ~fp32 accuracy at bf16 matrix-core rates.
// Double-buffered LDS + register prefetch: one barrier per K-step so the
// 256 MB HBM stream of U overlaps the WMMA issue. Single HBM pass over U
// (x-major grid -> the 8 column-tile siblings share U lines through L2).
// GCN scatter via global fp32 atomics; node-FC templated (no int division).
// ---------------------------------------------------------------------------

typedef __attribute__((ext_vector_type(16))) __bf16 v16bf;
typedef __attribute__((ext_vector_type(8)))  __bf16 v8bf;
typedef __attribute__((ext_vector_type(8)))  float  v8f;

union FragB { v16bf v; v8bf h[2]; };

__device__ __forceinline__ void splitf(float f, __bf16& h, __bf16& l) {
    h = (__bf16)f;                 // RNE truncation to bf16
    l = (__bf16)(f - (float)h);    // residual (Ah*Bl + Al*Bh recovers ~fp32)
}

// ---------------------------------------------------------------------------
// fc: latents[8,64] @ fc_W[64,16384] + fc_b -> feat0 stored [N=256][B=8][C=64]
// ---------------------------------------------------------------------------
__global__ __launch_bounds__(256) void fc_kernel(
    const float* __restrict__ lat, const float* __restrict__ W,
    const float* __restrict__ bias, float* __restrict__ out) {
    const int idx = blockIdx.x * 256 + threadIdx.x;   // < 256*8*64
    const int c = idx & 63;
    const int b = (idx >> 6) & 7;
    const int n = idx >> 9;
    const int col = n * 64 + c;
    float acc = bias[col];
    const float* lrow = lat + b * 64;
#pragma unroll 8
    for (int l = 0; l < 64; ++l) acc += lrow[l] * W[l * 16384 + col];
    out[(size_t)n * 512 + b * 64 + c] = acc;
}

// ---------------------------------------------------------------------------
// bf16x3 compensated GEMM: Y[M,N] = A[M,K] @ X[K,N] + Ub[row, col & cmask]
// block = 256 threads = 8 waves; block tile 128x32, K-step 32.
// Double-buffered LDS, register prefetch, one barrier per K-step.
// ---------------------------------------------------------------------------
#define MB 128
#define NB 32
#define KB 32
#define APAD 40   // A row stride in halves (80 B, 16B-aligned, bank-spread)
#define BPAD 48   // B row stride in halves (96 B)

__global__ __launch_bounds__(256) void gemm_bf16x3(
    const float* __restrict__ A, const float* __restrict__ X,
    const float* __restrict__ Ub, float* __restrict__ Y,
    int M, int K, int N, int C, int cmask) {
    __shared__ __bf16 Ah[2][MB * APAD];
    __shared__ __bf16 Al[2][MB * APAD];
    __shared__ __bf16 Bh[2][NB * BPAD];
    __shared__ __bf16 Bl[2][NB * BPAD];

    const int tid  = threadIdx.x;
    const int n0   = blockIdx.x * NB;     // x-major: siblings share U via L2
    const int m0   = blockIdx.y * MB;
    const int w    = tid >> 5;            // wave id -> 16-row M subtile
    const int lane = tid & 31;
    const int hi   = lane >> 4;           // lane group (ISA wave32 layouts)
    const int mn   = lane & 15;

    // per-thread staging coordinates
    const int a_col4 = (tid & 7) << 2;           // A: 4 chunks of row/col4
    const int b_krow = tid >> 3;                 // B: one float4
    float4 pa[4];
    float4 pb;

    auto g_load = [&](int k0) {
#pragma unroll
        for (int p = 0; p < 4; ++p) {
            const int row = (tid + p * 256) >> 3;     // 0..127
            pa[p] = *(const float4*)(A + (size_t)(m0 + row) * K + k0 + a_col4);
        }
        pb = *(const float4*)(X + (size_t)(k0 + b_krow) * N + n0 + a_col4);
    };

    auto l_store = [&](int bi) {
        __bf16* AhB = Ah[bi]; __bf16* AlB = Al[bi];
        __bf16* BhB = Bh[bi]; __bf16* BlB = Bl[bi];
        __bf16 h, l;
#pragma unroll
        for (int p = 0; p < 4; ++p) {
            const int row = (tid + p * 256) >> 3;
            splitf(pa[p].x, h, l); AhB[row*APAD + a_col4+0] = h; AlB[row*APAD + a_col4+0] = l;
            splitf(pa[p].y, h, l); AhB[row*APAD + a_col4+1] = h; AlB[row*APAD + a_col4+1] = l;
            splitf(pa[p].z, h, l); AhB[row*APAD + a_col4+2] = h; AlB[row*APAD + a_col4+2] = l;
            splitf(pa[p].w, h, l); AhB[row*APAD + a_col4+3] = h; AlB[row*APAD + a_col4+3] = l;
        }
        // B transposed to [n][k] so fragments are contiguous halves
        splitf(pb.x, h, l); BhB[(a_col4+0)*BPAD + b_krow] = h; BlB[(a_col4+0)*BPAD + b_krow] = l;
        splitf(pb.y, h, l); BhB[(a_col4+1)*BPAD + b_krow] = h; BlB[(a_col4+1)*BPAD + b_krow] = l;
        splitf(pb.z, h, l); BhB[(a_col4+2)*BPAD + b_krow] = h; BlB[(a_col4+2)*BPAD + b_krow] = l;
        splitf(pb.w, h, l); BhB[(a_col4+3)*BPAD + b_krow] = h; BlB[(a_col4+3)*BPAD + b_krow] = l;
    };

    v8f acc0 = {}; v8f acc1 = {};

    // pipeline prologue
    g_load(0);
    l_store(0);
    __syncthreads();
    if (K > KB) g_load(KB);

    int buf = 0;
    for (int k0 = 0;; k0 += KB) {
        // ---- compute from LDS[buf] ----
        const __bf16* AhB = Ah[buf]; const __bf16* AlB = Al[buf];
        const __bf16* BhB = Bh[buf]; const __bf16* BlB = Bl[buf];

        const int arow = w * 16 + mn;
        FragB ah, al, bh0, bl0, bh1, bl1;
        // A frag (16x32): lane m, halves = K[16*hi .. 16*hi+15]
        ah.h[0] = *(const v8bf*)&AhB[arow*APAD + 16*hi];
        ah.h[1] = *(const v8bf*)&AhB[arow*APAD + 16*hi + 8];
        al.h[0] = *(const v8bf*)&AlB[arow*APAD + 16*hi];
        al.h[1] = *(const v8bf*)&AlB[arow*APAD + 16*hi + 8];
        // B frag (32x16): lane n, halves = K[16*hi .. 16*hi+15]
        const int bn0 = mn, bn1 = 16 + mn;
        bh0.h[0] = *(const v8bf*)&BhB[bn0*BPAD + 16*hi];
        bh0.h[1] = *(const v8bf*)&BhB[bn0*BPAD + 16*hi + 8];
        bl0.h[0] = *(const v8bf*)&BlB[bn0*BPAD + 16*hi];
        bl0.h[1] = *(const v8bf*)&BlB[bn0*BPAD + 16*hi + 8];
        bh1.h[0] = *(const v8bf*)&BhB[bn1*BPAD + 16*hi];
        bh1.h[1] = *(const v8bf*)&BhB[bn1*BPAD + 16*hi + 8];
        bl1.h[0] = *(const v8bf*)&BlB[bn1*BPAD + 16*hi];
        bl1.h[1] = *(const v8bf*)&BlB[bn1*BPAD + 16*hi + 8];

        acc0 = __builtin_amdgcn_wmma_f32_16x16x32_bf16(false, ah.v, false, bh0.v, (short)0, acc0, false, false);
        acc0 = __builtin_amdgcn_wmma_f32_16x16x32_bf16(false, ah.v, false, bl0.v, (short)0, acc0, false, false);
        acc0 = __builtin_amdgcn_wmma_f32_16x16x32_bf16(false, al.v, false, bh0.v, (short)0, acc0, false, false);
        acc1 = __builtin_amdgcn_wmma_f32_16x16x32_bf16(false, ah.v, false, bh1.v, (short)0, acc1, false, false);
        acc1 = __builtin_amdgcn_wmma_f32_16x16x32_bf16(false, ah.v, false, bl1.v, (short)0, acc1, false, false);
        acc1 = __builtin_amdgcn_wmma_f32_16x16x32_bf16(false, al.v, false, bh1.v, (short)0, acc1, false, false);

        if (k0 + KB >= K) break;
        // ---- stage tile k0+KB into the other buffer, prefetch k0+2*KB ----
        l_store(buf ^ 1);
        __syncthreads();
        if (k0 + 2 * KB < K) g_load(k0 + 2 * KB);
        buf ^= 1;
    }

    // epilogue: C/D layout: VGPR r -> (M = r + 8*hi, N = lane&15)
#pragma unroll
    for (int r = 0; r < 8; ++r) {
        const int row = m0 + w * 16 + r + 8 * hi;
        const int c0  = n0 + mn;
        const int c1  = n0 + 16 + mn;
        Y[(size_t)row * N + c0] = acc0[r] + Ub[(size_t)row * C + (c0 & cmask)];
        Y[(size_t)row * N + c1] = acc1[r] + Ub[(size_t)row * C + (c1 & cmask)];
    }
}

// ---------------------------------------------------------------------------
// GCN pieces
// ---------------------------------------------------------------------------
__global__ void zero_kernel(float* __restrict__ p, long long n) {
    const long long i = (long long)blockIdx.x * 256 + threadIdx.x;
    if (i < n) p[i] = 0.0f;
}

// one block per edge; threads cover the B*C flattened channels (coalesced)
__global__ void scatter_kernel(const float* __restrict__ feat,
                               const int* __restrict__ rows,
                               const int* __restrict__ cols,
                               const float* __restrict__ vals,
                               float* __restrict__ agg, int BC) {
    const int e = blockIdx.x;
    const int j = threadIdx.x;
    const float v = vals[e];
    const int r = rows[e];
    const int c = cols[e];
    atomicAdd(&agg[(size_t)r * BC + j], v * feat[(size_t)c * BC + j]);
}

// out[o*so + b*sb + c] = relu(agg[o,b,:] @ W + bias); one thread per (o,b)
template <int Cin, int Cout>
__global__ void nodefc_kernel(const float* __restrict__ agg,
                              const float* __restrict__ W,
                              const float* __restrict__ bias,
                              float* __restrict__ out,
                              int M, long long so, long long sb) {
    const int idx = blockIdx.x * 256 + threadIdx.x;   // over M*8
    if (idx >= M * 8) return;
    const int b = idx & 7;
    const int o = idx >> 3;
    const float* a = agg + (size_t)o * (8 * Cin) + b * Cin;

    float acc[Cout];
#pragma unroll
    for (int c = 0; c < Cout; ++c) acc[c] = bias[c];
    for (int ci = 0; ci < Cin; ++ci) {
        const float av = a[ci];               // per-thread; W row is uniform
#pragma unroll
        for (int c = 0; c < Cout; ++c) acc[c] += av * W[ci * Cout + c];
    }
    float* op = out + (long long)o * so + (long long)b * sb;
#pragma unroll
    for (int c = 0; c < Cout; ++c) op[c] = fmaxf(acc[c], 0.0f);
}

// ---------------------------------------------------------------------------
extern "C" void kernel_launch(void* const* d_in, const int* in_sizes, int n_in,
                              void* d_out, int out_size, void* d_ws, size_t ws_size,
                              hipStream_t stream) {
    (void)in_sizes; (void)n_in; (void)out_size; (void)ws_size;

    const float* latents = (const float*)d_in[0];
    const float* fc_W    = (const float*)d_in[1];
    const float* fc_b    = (const float*)d_in[2];
    const float* U[3]    = {(const float*)d_in[3],  (const float*)d_in[10], (const float*)d_in[17]};
    const float* Ubp[3]  = {(const float*)d_in[4],  (const float*)d_in[11], (const float*)d_in[18]};
    const float* Wp[3]   = {(const float*)d_in[5],  (const float*)d_in[12], (const float*)d_in[19]};
    const float* bp[3]   = {(const float*)d_in[6],  (const float*)d_in[13], (const float*)d_in[20]};
    const float* vals[3] = {(const float*)d_in[7],  (const float*)d_in[14], (const float*)d_in[21]};
    const int*   rows[3] = {(const int*)d_in[8],    (const int*)d_in[15],   (const int*)d_in[22]};
    const int*   cols[3] = {(const int*)d_in[9],    (const int*)d_in[16],   (const int*)d_in[23]};

    // workspace carve (floats): buf0 <= 2M, buf1/buf2 <= 4.19M  (~42 MB total)
    float* buf0 = (float*)d_ws;
    float* buf1 = buf0 + 2097152;
    float* buf2 = buf1 + 4194304;

    static const int NSOUT[3] = {1024, 4096, 16384};
    static const int NSIN[3]  = {256, 1024, 4096};
    static const int CIN[3]   = {64, 64, 32};

    // fc -> feat0 [256][8][64]  (node-major, Ncols = B*C = 512)
    fc_kernel<<<(256 * 8 * 64) / 256, 256, 0, stream>>>(latents, fc_W, fc_b, buf0);

    const float* curIn = buf0;
    for (int i = 0; i < 3; ++i) {
        const int M = NSOUT[i], K = NSIN[i], C = CIN[i];
        const int Ncols = 8 * C;

        dim3 grid(Ncols / NB, M / MB);
        gemm_bf16x3<<<grid, 256, 0, stream>>>(U[i], curIn, Ubp[i], buf1,
                                              M, K, Ncols, C, C - 1);

        const long long aggN = (long long)M * Ncols;
        zero_kernel<<<(unsigned)((aggN + 255) / 256), 256, 0, stream>>>(buf2, aggN);

        const int E = M * 16;
        scatter_kernel<<<E, Ncols, 0, stream>>>(buf1, rows[i], cols[i], vals[i], buf2, Ncols);

        const unsigned nfcGrid = (unsigned)((M * 8 + 255) / 256);
        if (i == 0) {
            nodefc_kernel<64, 64><<<nfcGrid, 256, 0, stream>>>(
                buf2, Wp[i], bp[i], buf0, M, 8LL * 64, 64LL);   // [N][B][64]
            curIn = buf0;
        } else if (i == 1) {
            nodefc_kernel<64, 32><<<nfcGrid, 256, 0, stream>>>(
                buf2, Wp[i], bp[i], buf0, M, 8LL * 32, 32LL);   // [N][B][32]
            curIn = buf0;
        } else {
            nodefc_kernel<32, 3><<<nfcGrid, 256, 0, stream>>>(
                buf2, Wp[i], bp[i], (float*)d_out, M, 3LL, (long long)M * 3); // [B][16384][3]
        }
    }
}
